// GNN_GAT_11630771438170
// MI455X (gfx1250) — matrix operand order repaired and checked by reading, not compiled, not run on hardware
//
#include <hip/hip_runtime.h>

#define NNODES 50000
#define NEDGES 250000
#define NT 4           // edge types
#define NH 2           // heads
#define DDIM 64        // per-head dim
#define OUTC 128       // NH*DDIM
#define KPAD 64        // padded K for all layers
#define ROWT (NNODES/16)   // 3125 row tiles (exact)

typedef __attribute__((ext_vector_type(16))) _Float16 v16h;
typedef __attribute__((ext_vector_type(8)))  _Float16 v8h;
typedef __attribute__((ext_vector_type(8)))  float    v8f;

// ---------------- utility: zero fill ----------------
__global__ void k_zero(float* __restrict__ p, int n) {
    int i = blockIdx.x * blockDim.x + threadIdx.x;
    int stride = gridDim.x * blockDim.x;
    for (; i < n; i += stride) p[i] = 0.f;
}

// ---------------- pad/convert node features to f16 (K padded to 64) ----------------
__global__ void k_pad_x(const float* __restrict__ x, _Float16* __restrict__ xh, int ksrc) {
    int i = blockIdx.x * blockDim.x + threadIdx.x;   // over NNODES*KPAD
    if (i >= NNODES * KPAD) return;
    int n = i >> 6, k = i & 63;
    float v = (k < ksrc) ? x[(size_t)n * ksrc + k] : 0.f;
    xh[i] = (_Float16)v;
}

// ---------------- pack W (T,Ksrc,128) f32 -> WMMA B-fragments f16 ----------------
// frag f = t*16 + kt*8 + ct. B 32x16 f16 as packed here (k order matches A): lane L -> N = L&15,
// K = ktBase + (L>>4)*16 + h, halves h=0..15 stored K-ascending.
__global__ void k_pack_w(const float* __restrict__ W, _Float16* __restrict__ Bp, int ksrc) {
    int f = blockIdx.x;          // 0..63
    int lane = threadIdx.x;      // 0..31
    int t = f >> 4, kt = (f >> 3) & 1, ct = f & 7;
    int ncol  = ct * 16 + (lane & 15);
    int kbase = kt * 32 + ((lane >> 4) << 4);
    _Float16* dst = Bp + ((size_t)f * 32 + lane) * 16;
    for (int h = 0; h < 16; ++h) {
        int K = kbase + h;
        float v = (K < ksrc) ? W[((size_t)t * ksrc + K) * OUTC + ncol] : 0.f;
        dst[h] = (_Float16)v;
    }
}

// ---------------- WMMA GEMM: feat[t] = Ah (N x 64, f16) @ W[t] (64 x 128) ----------------
// One wave per 16x16 output tile. A 16x32 f16 as loaded here (k order matches B): lane L row M=L&15,
// halves 0-15 -> K = base..base+15, base = (L>>4)*16.
__global__ void __launch_bounds__(256) k_gemm(const _Float16* __restrict__ Ah,
                                              const _Float16* __restrict__ Bp,
                                              float* __restrict__ feat) {
    int wave = threadIdx.x >> 5;   // col tile 0..7
    int lane = threadIdx.x & 31;
    int rt = blockIdx.x;           // row tile
    int t  = blockIdx.y;           // etype
    int ct = wave;
    int row = rt * 16 + (lane & 15);

    v8f acc = {};
#pragma unroll
    for (int kt = 0; kt < 2; ++kt) {
        const _Float16* ab = Ah + (size_t)row * KPAD + kt * 32 + ((lane >> 4) << 4);
        v16h a = *(const v16h*)(ab);
        const v16h* bp = (const v16h*)(Bp + (((size_t)(t * 2 + kt) * 8 + ct) * 32 + lane) * 16);
        v16h b = *bp;
        acc = __builtin_amdgcn_wmma_f32_16x16x32_f16(false, a, false, b, (short)0, acc,
                                                     false, false);
    }
    // C/D 16x16 f32: VGPR r -> M = r + (lane>=16 ? 8 : 0), N = lane&15
    int colBase = ct * 16 + (lane & 15);
    int rowHi = (lane >> 4) << 3;
    float* cb = feat + ((size_t)t * NNODES + (size_t)rt * 16) * OUTC + colBase;
#pragma unroll
    for (int r = 0; r < 8; ++r)
        cb[(size_t)(rowHi + r) * OUTC] = acc[r];
}

// ---------------- el/er attention dots ----------------
__global__ void k_elr(const float* __restrict__ feat, const float* __restrict__ al,
                      const float* __restrict__ ar, float* __restrict__ el,
                      float* __restrict__ er) {
    int n = blockIdx.x * blockDim.x + threadIdx.x;
    int t = blockIdx.y;
    if (n >= NNODES) return;
    const float* f = feat + ((size_t)t * NNODES + n) * OUTC;
    for (int h = 0; h < NH; ++h) {
        const float* a = al + (size_t)(t * NH + h) * DDIM;
        const float* r = ar + (size_t)(t * NH + h) * DDIM;
        float sl = 0.f, sr = 0.f;
        for (int k = 0; k < DDIM; ++k) {
            float fv = f[h * DDIM + k];
            sl += fv * a[k];
            sr += fv * r[k];
        }
        el[((size_t)t * NNODES + n) * NH + h] = sl;
        er[((size_t)t * NNODES + n) * NH + h] = sr;
    }
}

// ---------------- edge pass 1: leaky-relu, exp, segment-sum denom ----------------
// (softmax max-subtraction skipped: alpha is exactly invariant and logits are tiny)
__global__ void k_edge_att(const int* __restrict__ edges, const float* __restrict__ el,
                           const float* __restrict__ er, float* __restrict__ denom,
                           float* __restrict__ exv) {
    int e = blockIdx.x * blockDim.x + threadIdx.x;
    int t = blockIdx.y;
    if (e >= NEDGES) return;
    int s = edges[((size_t)t * 2 + 0) * NEDGES + e];
    int d = edges[((size_t)t * 2 + 1) * NEDGES + e];
    for (int h = 0; h < NH; ++h) {
        float v = el[((size_t)t * NNODES + s) * NH + h]
                + er[((size_t)t * NNODES + d) * NH + h];
        v = v > 0.f ? v : 0.2f * v;
        float ex = expf(v);
        exv[((size_t)t * NEDGES + e) * NH + h] = ex;
        atomicAdd(&denom[((size_t)t * NNODES + d) * NH + h], ex);
    }
}

// ---------------- edge pass 2: gather feat[src]*alpha, scatter-add to out[dst] ----------------
// 4 threads per edge, each handles 32 contiguous cols (one head half).
__global__ void k_edge_msg(const int* __restrict__ edges, const float* __restrict__ feat,
                           const float* __restrict__ exv, const float* __restrict__ denom,
                           float* __restrict__ outacc) {
    int idx = blockIdx.x * blockDim.x + threadIdx.x;   // over NEDGES*4
    int t = blockIdx.y;
    if (idx >= NEDGES * 4) return;
    int e = idx >> 2, q = idx & 3;
    int s = edges[((size_t)t * 2 + 0) * NEDGES + e];
    int d = edges[((size_t)t * 2 + 1) * NEDGES + e];
    int h = q >> 1;
    float alpha = exv[((size_t)t * NEDGES + e) * NH + h]
                / denom[((size_t)t * NNODES + d) * NH + h];
    const float4* fs = (const float4*)(feat + ((size_t)t * NNODES + s) * OUTC + q * 32);
    float* od = outacc + ((size_t)t * NNODES + d) * OUTC + q * 32;
#pragma unroll
    for (int i = 0; i < 8; ++i) {
        float4 v = fs[i];
        atomicAdd(od + 4 * i + 0, v.x * alpha);
        atomicAdd(od + 4 * i + 1, v.y * alpha);
        atomicAdd(od + 4 * i + 2, v.z * alpha);
        atomicAdd(od + 4 * i + 3, v.w * alpha);
    }
}

// ---------------- combine: mean over etypes + bias, relu, head-normalize, head-mean ----------------
__global__ void k_combine(const float* __restrict__ outacc, const float* __restrict__ b,
                          _Float16* __restrict__ hnext) {
    int i = blockIdx.x * blockDim.x + threadIdx.x;   // over NNODES*DDIM
    if (i >= NNODES * DDIM) return;
    int n = i >> 6, dd = i & 63;
    float v0 = 0.f, v1 = 0.f;
    for (int t = 0; t < NT; ++t) {
        v0 += outacc[((size_t)t * NNODES + n) * OUTC + dd]        + b[(t * NH + 0) * DDIM + dd];
        v1 += outacc[((size_t)t * NNODES + n) * OUTC + DDIM + dd] + b[(t * NH + 1) * DDIM + dd];
    }
    v0 *= 0.25f; v1 *= 0.25f;
    v0 = fmaxf(v0, 0.f); v1 = fmaxf(v1, 0.f);
    float norm = sqrtf(v0 * v0 + v1 * v1);
    float inv = 1.f / fmaxf(norm, 1e-12f);
    float m = 0.5f * (v0 + v1) * inv;
    hnext[(size_t)n * KPAD + dd] = (_Float16)m;
}

// ---------------- final: mean over etypes + heads, then deterministic node-mean ----------------
__global__ void k_reduce1(const float* __restrict__ outacc, const float* __restrict__ b,
                          float* __restrict__ partial) {
    int dd = threadIdx.x;    // 64
    int g  = blockIdx.x;     // 128
    float bs = 0.f;
    for (int t = 0; t < NT; ++t)
        bs += b[(t * NH + 0) * DDIM + dd] + b[(t * NH + 1) * DDIM + dd];
    float sum = 0.f;
    for (int n = g; n < NNODES; n += 128) {
        float o = 0.f;
        for (int t = 0; t < NT; ++t) {
            o += outacc[((size_t)t * NNODES + n) * OUTC + dd];
            o += outacc[((size_t)t * NNODES + n) * OUTC + DDIM + dd];
        }
        sum += 0.125f * (o + bs);   // mean over 4 types, mean over 2 heads
    }
    partial[g * 64 + dd] = sum;
}

__global__ void k_reduce2(const float* __restrict__ partial, float* __restrict__ out) {
    int dd = threadIdx.x;
    float s = 0.f;
    for (int g = 0; g < 128; ++g) s += partial[g * 64 + dd];
    out[dd] = s * (1.f / (float)NNODES);
}

// ---------------- host launch ----------------
extern "C" void kernel_launch(void* const* d_in, const int* in_sizes, int n_in,
                              void* d_out, int out_size, void* d_ws, size_t ws_size,
                              hipStream_t stream) {
    (void)in_sizes; (void)n_in; (void)out_size; (void)ws_size;
    const float* x     = (const float*)d_in[0];
    const int*   edges = (const int*)d_in[1];
    const float* W[3]  = {(const float*)d_in[2], (const float*)d_in[6],  (const float*)d_in[10]};
    const float* al[3] = {(const float*)d_in[3], (const float*)d_in[7],  (const float*)d_in[11]};
    const float* ar[3] = {(const float*)d_in[4], (const float*)d_in[8],  (const float*)d_in[12]};
    const float* bb[3] = {(const float*)d_in[5], (const float*)d_in[9],  (const float*)d_in[13]};
    const int Ksrc[3] = {23, 64, 64};

    char* ws = (char*)d_ws;
    size_t off = 0;
    auto alloc = [&](size_t bytes) -> void* {
        void* p = ws + off;
        off = (off + bytes + 255) & ~(size_t)255;
        return p;
    };
    _Float16* xh   = (_Float16*)alloc((size_t)NNODES * KPAD * 2);         // 6.4 MB
    _Float16* Bp   = (_Float16*)alloc((size_t)64 * 32 * 16 * 2);          // 64 KB
    float* feat    = (float*)alloc((size_t)NT * NNODES * OUTC * 4);       // 102.4 MB
    float* el      = (float*)alloc((size_t)NT * NNODES * NH * 4);
    float* er      = (float*)alloc((size_t)NT * NNODES * NH * 4);
    float* denom   = (float*)alloc((size_t)NT * NNODES * NH * 4);
    float* exv     = (float*)alloc((size_t)NT * NEDGES * NH * 4);         // 8 MB
    float* outacc  = (float*)alloc((size_t)NT * NNODES * OUTC * 4);       // 102.4 MB
    float* partial = (float*)alloc((size_t)128 * 64 * 4);

    k_pad_x<<<(NNODES * KPAD + 255) / 256, 256, 0, stream>>>(x, xh, 23);

    for (int l = 0; l < 3; ++l) {
        k_pack_w<<<64, 32, 0, stream>>>(W[l], Bp, Ksrc[l]);
        k_gemm<<<dim3(ROWT, NT), 256, 0, stream>>>(xh, Bp, feat);
        k_elr<<<dim3((NNODES + 255) / 256, NT), 256, 0, stream>>>(feat, al[l], ar[l], el, er);
        k_zero<<<1024, 256, 0, stream>>>(denom, NT * NNODES * NH);
        k_zero<<<4096, 256, 0, stream>>>(outacc, NT * NNODES * OUTC);
        k_edge_att<<<dim3((NEDGES + 255) / 256, NT), 256, 0, stream>>>(edges, el, er, denom, exv);
        k_edge_msg<<<dim3((NEDGES * 4 + 255) / 256, NT), 256, 0, stream>>>(edges, feat, exv,
                                                                           denom, outacc);
        if (l < 2) {
            k_combine<<<(NNODES * DDIM + 255) / 256, 256, 0, stream>>>(outacc, bb[l], xh);
        } else {
            k_reduce1<<<128, 64, 0, stream>>>(outacc, bb[l], partial);
            k_reduce2<<<1, 64, 0, stream>>>(partial, (float*)d_out);
        }
    }
}
